// MambaBlock_74474732912932
// MI455X (gfx1250) — matrix operand order repaired and checked by reading, not compiled, not run on hardware
//
#include <hip/hip_runtime.h>
#include <hip/hip_bf16.h>

typedef __attribute__((ext_vector_type(16))) __bf16 v16bf;
typedef __attribute__((ext_vector_type(8)))  __bf16 v8bf;
typedef __attribute__((ext_vector_type(4)))  __bf16 v4bf;
typedef __attribute__((ext_vector_type(8)))  float  v8f;

#define D_MODEL 1024
#define D_STATE 16
#define D_CONV  4
#define DT_RANK 64
#define D_INNER 2048
#define B_SZ    2
#define L_SEQ   2048
#define M_ROWS  (B_SZ * L_SEQ)      // 4096
#define LDS_STRIDE 40               // elems per LDS tile row (32 data + 8 pad)

static __device__ __forceinline__ float sigmoid_f(float x) { return 1.0f / (1.0f + __expf(-x)); }
static __device__ __forceinline__ float silu_f(float x)    { return x * sigmoid_f(x); }

// Async global->LDS 16B copy (GVS form): vdst = LDS byte address, vaddr = byte
// offset, saddr = 64-bit base. Tracked by ASYNCcnt.
static __device__ __forceinline__ void async_copy_b128(unsigned ldsAddr,
                                                       unsigned gOff,
                                                       unsigned long long gBase) {
    asm volatile("global_load_async_to_lds_b128 %0, %1, %2"
                 :: "v"(ldsAddr), "v"(gOff), "s"(gBase) : "memory");
}
static __device__ __forceinline__ void wait_async_zero() {
    asm volatile("s_wait_asynccnt 0" ::: "memory");
}
// Low 32 bits of a generic pointer to LDS == LDS byte offset (flat aperture rule).
template <typename T>
static __device__ __forceinline__ unsigned lds_addr_of(const T* p) {
    return (unsigned)(unsigned long long)p;
}

// Elementwise f32 -> bf16 (RNE via v_cvt_pk_bf16_f32), 4 elems/thread. n % 4 == 0.
__global__ __launch_bounds__(256)
void f32_to_bf16_kernel(const float* __restrict__ src, __bf16* __restrict__ dst, int n) {
    int base = (blockIdx.x * 256 + threadIdx.x) * 4;
    if (base >= n) return;
    float4 v = *(const float4*)(src + base);
    v4bf o = { (__bf16)v.x, (__bf16)v.y, (__bf16)v.z, (__bf16)v.w };
    *(v4bf*)(dst + base) = o;
}

// C[m,n] = act( sum_k A[m,k] * W[n,k] + bias[n] )
// A: f32 (M,K) stride lda ; Wb: bf16 (N,K) stride ldw ; C f32 stride ldc.
// Block tile 128(M) x 64(N), K-step 32, 256 threads = 8 waves (wave32),
// each wave owns a 32x32 subtile = 2x2 WMMA_F32_16X16X32_BF16 accumulators.
// Software-pipelined: double-buffered LDS, slab i+1 staged with
// GLOBAL_LOAD_ASYNC_TO_LDS_B128 while slab i is consumed; each wave drains its
// ASYNCcnt to 0 before signaling the single per-iteration barrier.
// act: 0 = identity, 1 = softplus.
__global__ __launch_bounds__(256)
void wmma_gemm_bias_act(const float* __restrict__ A, int lda,
                        const __bf16* __restrict__ Wb, int ldw,
                        const float* __restrict__ bias,
                        float* __restrict__ C, int ldc,
                        int M, int N, int K, int act)
{
    __shared__ __align__(16) float  lAf[2][128 * LDS_STRIDE];  // 2 x 20 KB (f32 tiles)
    __shared__ __align__(16) __bf16 lWb[2][64  * LDS_STRIDE];  // 2 x 5 KB  (bf16 tiles)

    const int tid     = threadIdx.x;
    const int lane    = tid & 31;
    const int wave    = tid >> 5;
    const int waveRow = wave >> 1;            // 0..3  (M direction, 32 rows each)
    const int waveCol = wave & 1;             // 0..1  (N direction, 32 cols each)
    const int mBase   = blockIdx.y * 128;
    const int nBase   = blockIdx.x * 64;
    const int l16     = lane & 15;
    const int halfSel = (lane >> 4) & 1;      // lanes 0-15 vs 16-31

    const unsigned long long Abase = (unsigned long long)A;
    const unsigned long long Wbase = (unsigned long long)Wb;

    // Per-thread staging coordinates (invariant over K loop).
    const int aRow  = tid >> 3;               // A: 128 rows x 8 chunks of 16B
    const int aC4   = (tid & 7) << 2;         //    f32 column of this 16B chunk
    const int wRowS = tid >> 2;               // W: 64 rows x 4 chunks of 16B
    const int wSeg  = (tid & 3) << 3;         //    bf16 column of this 16B chunk

    auto stage = [&](int buf, int kk) {
        #pragma unroll
        for (int t = 0; t < 4; ++t) {
            int row = aRow + t * 32;
            unsigned lo = lds_addr_of(&lAf[buf][row * LDS_STRIDE + aC4]);
            unsigned go = (unsigned)((((size_t)(mBase + row)) * lda + kk + aC4) * 4u);
            async_copy_b128(lo, go, Abase);
        }
        unsigned lo = lds_addr_of(&lWb[buf][wRowS * LDS_STRIDE + wSeg]);
        unsigned go = (unsigned)((((size_t)(nBase + wRowS)) * ldw + kk + wSeg) * 2u);
        async_copy_b128(lo, go, Wbase);
    };

    const v8f vzero = {0.f,0.f,0.f,0.f,0.f,0.f,0.f,0.f};
    v8f acc[2][2];
    acc[0][0] = vzero; acc[0][1] = vzero; acc[1][0] = vzero; acc[1][1] = vzero;

    stage(0, 0);                               // prologue: slab 0 in flight

    for (int kk = 0; kk < K; kk += 32) {
        const int cur = (kk >> 5) & 1;

        // Drain own async copies of slab `cur`, then block-wide rendezvous:
        // after this barrier every wave's slab-`cur` bytes are in LDS, and the
        // previous consumers of buffer cur^1 are provably done with it.
        wait_async_zero();
        __syncthreads();

        // Kick off next slab into the other buffer; overlaps with compute below.
        if (kk + 32 < K) stage(cur ^ 1, kk + 32);
        if (kk + 64 < K)                       // L2 prefetch two slabs ahead
            __builtin_prefetch(A + (size_t)(mBase + (tid >> 1)) * lda + kk + 64, 0, 1);

        // ---- build fragments per ISA per-lane layouts ----
        // A (16x32 bf16): lane row M=l16; lanes<16 hold K {0..7,16..23}, lanes>=16 {8..15,24..31}
        const int aK = halfSel * 8;
        // B (32x16 bf16): lane col N=l16; lanes<16 hold K {0..7,16..23}, lanes>=16 {8..15,24..31}
        const int bK = halfSel * 8;

        v16bf afrag[2], bfrag[2];
        #pragma unroll
        for (int mg = 0; mg < 2; ++mg) {
            const float* pa = &lAf[cur][(waveRow * 32 + mg * 16 + l16) * LDS_STRIDE + aK];
            float tmp[16];
            *(float4*)&tmp[0]  = *(const float4*)(pa);
            *(float4*)&tmp[4]  = *(const float4*)(pa + 4);
            *(float4*)&tmp[8]  = *(const float4*)(pa + 16);
            *(float4*)&tmp[12] = *(const float4*)(pa + 20);
            v16bf af;
            #pragma unroll
            for (int e = 0; e < 16; ++e) af[e] = (__bf16)tmp[e];
            afrag[mg] = af;
        }
        #pragma unroll
        for (int ng = 0; ng < 2; ++ng) {
            const __bf16* pw = &lWb[cur][(waveCol * 32 + ng * 16 + l16) * LDS_STRIDE + bK];
            v8bf lo = *(const v8bf*)(pw);
            v8bf hi = *(const v8bf*)(pw + 16);
            bfrag[ng] = __builtin_shufflevector(lo, hi, 0,1,2,3,4,5,6,7,8,9,10,11,12,13,14,15);
        }

        #pragma unroll
        for (int mg = 0; mg < 2; ++mg)
            #pragma unroll
            for (int ng = 0; ng < 2; ++ng)
                acc[mg][ng] = __builtin_amdgcn_wmma_f32_16x16x32_bf16(
                    false, afrag[mg], false, bfrag[ng],
                    (short)0, acc[mg][ng], false, false);
    }

    // ---- epilogue: C/D layout: elem j -> (M = j + 8*halfSel, N = l16) within 16x16 ----
    const int mOffLane = halfSel * 8;
    #pragma unroll
    for (int mg = 0; mg < 2; ++mg) {
        #pragma unroll
        for (int ng = 0; ng < 2; ++ng) {
            int col = nBase + waveCol * 32 + ng * 16 + l16;
            if (col >= N) continue;
            float bv = bias[col];
            #pragma unroll
            for (int j = 0; j < 8; ++j) {
                int row = mBase + waveRow * 32 + mg * 16 + mOffLane + j;
                float v = acc[mg][ng][j] + bv;
                if (act == 1) v = (v > 20.0f) ? v : __logf(1.0f + __expf(v));
                C[(size_t)row * ldc + col] = v;
            }
        }
    }
}

// Causal depthwise conv (width 4) over L + SiLU. x_inner lives in cols [0,2048) of xz.
__global__ __launch_bounds__(256)
void conv_silu_kernel(const float* __restrict__ xz, const float* __restrict__ conv_w,
                      const float* __restrict__ conv_b, float* __restrict__ xc)
{
    int idx = blockIdx.x * 256 + threadIdx.x;        // (b, l, d), d fastest
    if (idx >= B_SZ * L_SEQ * D_INNER) return;
    int d = idx & (D_INNER - 1);
    int l = (idx >> 11) & (L_SEQ - 1);
    int b = idx >> 22;
    float acc = conv_b[d];
    #pragma unroll
    for (int j = 0; j < D_CONV; ++j) {
        int ll = l - (D_CONV - 1) + j;
        if (ll >= 0)
            acc += conv_w[d * D_CONV + j] *
                   xz[(size_t)(b * L_SEQ + ll) * (2 * D_INNER) + d];
    }
    xc[idx] = silu_f(acc);
}

// Selective scan: one thread per (b, d) channel; 16-state recurrence in registers.
// Fuses y = scan + D*x_conv, then y *= silu(z). Writes gated y into xz cols [0,2048)
// (x_inner half, dead after conv) and h_last to the output tail.
__global__ __launch_bounds__(256)
void scan_kernel(const float* __restrict__ xc, const float* __restrict__ dt,
                 const float* __restrict__ xdbl, float* __restrict__ xz,
                 const float* __restrict__ A_log, const float* __restrict__ D_param,
                 float* __restrict__ h_last)
{
    int ch = blockIdx.x * 256 + threadIdx.x;         // 0 .. B*D_INNER-1
    if (ch >= B_SZ * D_INNER) return;
    int b = ch >> 11;
    int d = ch & (D_INNER - 1);

    float Aneg[D_STATE];
    #pragma unroll
    for (int s = 0; s < D_STATE; ++s) Aneg[s] = -__expf(A_log[d * D_STATE + s]);
    const float Dp = D_param[d];

    float h[D_STATE];
    #pragma unroll
    for (int s = 0; s < D_STATE; ++s) h[s] = 0.0f;

    for (int t = 0; t < L_SEQ; ++t) {
        size_t r = (size_t)(b * L_SEQ + t);
        float dtv = dt[r * D_INNER + d];
        float xv  = xc[r * D_INNER + d];
        float zv  = xz[r * (2 * D_INNER) + D_INNER + d];

        const float* sel = xdbl + r * (DT_RANK + 2 * D_STATE);
        float Bs[D_STATE], Cs[D_STATE];
        #pragma unroll
        for (int q = 0; q < 4; ++q) {
            *(float4*)&Bs[q * 4] = *(const float4*)(sel + DT_RANK + q * 4);
            *(float4*)&Cs[q * 4] = *(const float4*)(sel + DT_RANK + D_STATE + q * 4);
        }

        float dx = dtv * xv;
        float y  = 0.0f;
        #pragma unroll
        for (int s = 0; s < D_STATE; ++s) {
            h[s] = h[s] * __expf(dtv * Aneg[s]) + dx * Bs[s];
            y += h[s] * Cs[s];
        }
        float yo = (y + Dp * xv) * silu_f(zv);
        xz[r * (2 * D_INNER) + d] = yo;
    }
    #pragma unroll
    for (int s = 0; s < D_STATE; ++s)
        h_last[(size_t)ch * D_STATE + s] = h[s];
}

extern "C" void kernel_launch(void* const* d_in, const int* in_sizes, int n_in,
                              void* d_out, int out_size, void* d_ws, size_t ws_size,
                              hipStream_t stream) {
    const float* x       = (const float*)d_in[0];
    const float* W_in    = (const float*)d_in[1];
    const float* b_in    = (const float*)d_in[2];
    const float* conv_w  = (const float*)d_in[3];
    const float* conv_b  = (const float*)d_in[4];
    const float* W_x     = (const float*)d_in[5];
    const float* b_x     = (const float*)d_in[6];
    const float* W_dt    = (const float*)d_in[7];
    const float* b_dt    = (const float*)d_in[8];
    const float* A_log   = (const float*)d_in[9];
    const float* D_param = (const float*)d_in[10];
    const float* W_out   = (const float*)d_in[11];
    const float* b_out   = (const float*)d_in[12];

    float* out   = (float*)d_out;                               // (B, L, D_MODEL)
    float* hlast = out + (size_t)M_ROWS * D_MODEL;              // (B, D_INNER, D_STATE)

    // workspace layout
    float* ws    = (float*)d_ws;
    float* xz    = ws;                                          // (4096, 4096) f32
    float* xc    = xz   + (size_t)M_ROWS * (2 * D_INNER);       // (4096, 2048) f32
    float* xdbl  = xc   + (size_t)M_ROWS * D_INNER;             // (4096, 96)   f32
    float* dtbuf = xdbl + (size_t)M_ROWS * (DT_RANK + 2 * D_STATE); // (4096, 2048) f32
    __bf16* Win_b  = (__bf16*)(dtbuf + (size_t)M_ROWS * D_INNER);
    __bf16* Wx_b   = Win_b + (size_t)(2 * D_INNER) * D_MODEL;
    __bf16* Wdt_b  = Wx_b  + (size_t)(DT_RANK + 2 * D_STATE) * D_INNER;
    __bf16* Wout_b = Wdt_b + (size_t)D_INNER * DT_RANK;

    dim3 blk(256);

    // 0) one-time (per launch) weight conversions f32 -> bf16
    {
        int n1 = (2 * D_INNER) * D_MODEL;
        int n2 = (DT_RANK + 2 * D_STATE) * D_INNER;
        int n3 = D_INNER * DT_RANK;
        int n4 = D_MODEL * D_INNER;
        f32_to_bf16_kernel<<<n1 / 1024, blk, 0, stream>>>(W_in,  Win_b,  n1);
        f32_to_bf16_kernel<<<n2 / 1024, blk, 0, stream>>>(W_x,   Wx_b,   n2);
        f32_to_bf16_kernel<<<n3 / 1024, blk, 0, stream>>>(W_dt,  Wdt_b,  n3);
        f32_to_bf16_kernel<<<n4 / 1024, blk, 0, stream>>>(W_out, Wout_b, n4);
    }

    // 1) xz = x @ W_in^T + b_in          (M=4096, N=4096, K=1024)
    wmma_gemm_bias_act<<<dim3((2 * D_INNER) / 64, M_ROWS / 128), blk, 0, stream>>>(
        x, D_MODEL, Win_b, D_MODEL, b_in, xz, 2 * D_INNER,
        M_ROWS, 2 * D_INNER, D_MODEL, 0);

    // 2) x_conv = silu(depthwise_causal_conv(x_inner))
    conv_silu_kernel<<<(B_SZ * L_SEQ * D_INNER) / 256, blk, 0, stream>>>(
        xz, conv_w, conv_b, xc);

    // 3) x_dbl = x_conv @ W_x^T + b_x    (M=4096, N=96, K=2048)
    wmma_gemm_bias_act<<<dim3(2, M_ROWS / 128), blk, 0, stream>>>(
        xc, D_INNER, Wx_b, D_INNER, b_x, xdbl, DT_RANK + 2 * D_STATE,
        M_ROWS, DT_RANK + 2 * D_STATE, D_INNER, 0);

    // 4) dt = softplus(dt_low @ W_dt^T + b_dt)   (M=4096, N=2048, K=64)
    wmma_gemm_bias_act<<<dim3(D_INNER / 64, M_ROWS / 128), blk, 0, stream>>>(
        xdbl, DT_RANK + 2 * D_STATE, Wdt_b, DT_RANK, b_dt, dtbuf, D_INNER,
        M_ROWS, D_INNER, DT_RANK, 1);

    // 5) selective scan + D-skip + z-gating; gated y overwrites x_inner half of xz
    scan_kernel<<<(B_SZ * D_INNER) / 256, blk, 0, stream>>>(
        xc, dtbuf, xdbl, xz, A_log, D_param, hlast);

    // 6) out = y @ W_out^T + b_out       (M=4096, N=1024, K=2048), A = xz (lda=4096)
    wmma_gemm_bias_act<<<dim3(D_MODEL / 64, M_ROWS / 128), blk, 0, stream>>>(
        xz, 2 * D_INNER, Wout_b, D_INNER, b_out, out, D_MODEL,
        M_ROWS, D_MODEL, D_INNER, 0);
}
